// TrajectoryAttention_4157528343340
// MI455X (gfx1250) — matrix operand order repaired and checked by reading, not compiled, not run on hardware
//
#include <hip/hip_runtime.h>
#include <hip/hip_bf16.h>
#include <stdint.h>
#include <stddef.h>

typedef __attribute__((ext_vector_type(16))) _Float16 v16h;
typedef __attribute__((ext_vector_type(8)))  _Float16 v8h;
typedef __attribute__((ext_vector_type(8)))  float    v8f;
typedef __attribute__((ext_vector_type(4)))  int      v4i;

#define DEV __device__ __forceinline__

// Optional CDNA5 async global->LDS path (compile-time probed).
#if defined(__has_builtin)
#if __has_builtin(__builtin_amdgcn_global_load_async_to_lds_b128) && \
    __has_builtin(__builtin_amdgcn_s_wait_asynccnt)
#define HAVE_ASYNC 1
#endif
#endif
#ifndef HAVE_ASYNC
#define HAVE_ASYNC 0
#endif

#define AS1 __attribute__((address_space(1)))
#define AS3 __attribute__((address_space(3)))

// ---- problem constants (from setup_inputs) ----
constexpr int kB  = 2;
constexpr int kH  = 12;
constexpr int kBH = 24;      // B*H
constexpr int kF  = 16;      // frames
constexpr int kP  = 196;     // tokens per frame
constexpr int kN  = 3136;    // F*P
constexpr int kC  = 768;
constexpr int kD  = 64;      // head dim
constexpr int kL  = 128;     // landmarks
constexpr int kC3 = 2304;    // 3C
constexpr int kC2 = 1536;    // 2C
constexpr int kM  = kB * kN; // 6272 rows
constexpr int kFD = kF * kD; // 1024
constexpr float kScQK = 0.35355339059327379f;  // d^-0.25
constexpr float kScAt = 0.125f;                // d^-0.5

struct IdxArr { int v[kBH]; };

// ---------------- WMMA helpers (CDNA5 16x16x32 f16 -> f32) ----------------
DEV v8f wmma_f16(v16h a, v16h b, v8f c) {
  return __builtin_amdgcn_wmma_f32_16x16x32_f16(false, a, false, b, (short)0, c,
                                                false, false);
}

DEV v16h cvt16(const float* t, float scale) {
  v16h r;
#pragma unroll
  for (int j = 0; j < 16; ++j) r[j] = (_Float16)(t[j] * scale);
  return r;
}

// A fragment from f16 row-major (global or LDS), fully aligned fast path.
// lane<16 -> M=lane, K kBase+0..15; lane>=16 -> K kBase+16..31.
DEV v16h load_a_h(const _Float16* A, int lda, int mBase, int kBase, int lane) {
  const _Float16* row = A + (size_t)(mBase + (lane & 15)) * lda + kBase +
                        ((lane >= 16) ? 16 : 0);
  v8h lo = *(const v8h*)row;
  v8h hi = *(const v8h*)(row + 8);
  return __builtin_shufflevector(lo, hi, 0, 1, 2, 3, 4, 5, 6, 7, 8, 9, 10, 11,
                                 12, 13, 14, 15);
}

// A fragment from f16 row-major, scalar (handles misaligned rows / K tail).
template <bool CHKK>
DEV v16h load_a_hs(const _Float16* A, int lda, int mBase, int kBase, int lane,
                   int kMax) {
  const _Float16* row = A + (size_t)(mBase + (lane & 15)) * lda;
  const int kOff = (lane >= 16) ? 16 : 0;
  v16h r;
#pragma unroll
  for (int j = 0; j < 16; ++j) {
    int k = kBase + kOff + j;
    r[j] = CHKK ? ((k < kMax) ? row[k] : (_Float16)0.f) : row[k];
  }
  return r;
}

// A fragment from f32 row-major via float4 loads (aligned cases only).
DEV v16h load_a_f32v(const float* __restrict__ A, int lda, int mBase, int kBase,
                     int lane, float scale) {
  const float* row = A + (size_t)(mBase + (lane & 15)) * lda + kBase +
                     ((lane >= 16) ? 16 : 0);
  float t[16];
  *(float4*)(t + 0)  = *(const float4*)(row + 0);
  *(float4*)(t + 4)  = *(const float4*)(row + 4);
  *(float4*)(t + 8)  = *(const float4*)(row + 8);
  *(float4*)(t + 12) = *(const float4*)(row + 12);
  return cvt16(t, scale);
}

// B fragment from pre-transposed f16 weights Wt[n][k] (stride ldk): one
// contiguous 32B vector load per lane.
// lane<16 -> N=nBase+lane, K=kBase+0..15 ; lane>=16 -> K=kBase+16..31.
DEV v16h load_bth(const _Float16* Wt, int ldk, int kBase, int nBase, int lane) {
  return *(const v16h*)(Wt + (size_t)(nBase + (lane & 15)) * ldk + kBase +
                        ((lane >= 16) ? 16 : 0));
}

// B fragment from transposed f32 view M[n][k] via float4 loads.
DEV v16h load_bt_f32v(const float* __restrict__ Mp, int ldm, int kBase,
                      int nBase, int lane, float scale) {
  const float* row = Mp + (size_t)(nBase + (lane & 15)) * ldm + kBase +
                     ((lane >= 16) ? 16 : 0);
  float t[16];
  *(float4*)(t + 0)  = *(const float4*)(row + 0);
  *(float4*)(t + 4)  = *(const float4*)(row + 4);
  *(float4*)(t + 8)  = *(const float4*)(row + 8);
  *(float4*)(t + 12) = *(const float4*)(row + 12);
  return cvt16(t, scale);
}

// B fragment from row-major f32 B[k][n] (column access, scalar).
template <bool CHKK>
DEV v16h load_b(const float* __restrict__ Bp, int ldb, int kBase, int nBase,
                int lane, float scale, int kMax) {
  const int n = nBase + (lane & 15);
  const int kb = kBase + ((lane >= 16) ? 16 : 0);
  v16h r;
#pragma unroll
  for (int j = 0; j < 16; ++j) {
    int k = kb + j;
    float v = CHKK ? ((k < kMax) ? Bp[(size_t)k * ldb + n] : 0.0f)
                   : Bp[(size_t)k * ldb + n];
    r[j] = (_Float16)(v * scale);
  }
  return r;
}

// C/D store: VGPR v -> M = mBase+v (lanes 0-15) or mBase+v+8 (lanes 16-31).
DEV void store_c(float* __restrict__ Cp, int ldc, int mBase, int nBase,
                 int lane, v8f acc) {
  const int n = nBase + (lane & 15);
  const int mAdd = (lane >= 16) ? 8 : 0;
#pragma unroll
  for (int v = 0; v < 8; ++v)
    Cp[(size_t)(mBase + v + mAdd) * ldc + n] = acc[v];
}

DEV float wred_max(float v) {
  for (int o = 16; o > 0; o >>= 1) v = fmaxf(v, __shfl_xor(v, o, 32));
  return v;
}
DEV float wred_sum(float v) {
  for (int o = 16; o > 0; o >>= 1) v += __shfl_xor(v, o, 32);
  return v;
}

// ============ 0) weight convert+transpose: Wt[n][k] = (f16)W[k][n] ============
__global__ __launch_bounds__(256) void k_wt_cvt(const float* __restrict__ W,
                                                _Float16* __restrict__ Wt,
                                                int K, int Nc) {
  int idx = blockIdx.x * 256 + threadIdx.x;
  if (idx >= K * Nc) return;
  int n = idx / K, k = idx - n * K;
  Wt[idx] = (_Float16)W[(size_t)k * Nc + n];
}

// ============ 1) qkv = x @ W_qkv : A tile staged once in LDS (f16) ============
__global__ __launch_bounds__(256) void k_gemm_qkv(const float* __restrict__ X,
                                                  const _Float16* __restrict__ WqkvT,
                                                  float* __restrict__ O) {
  __shared__ _Float16 xa[16 * kC];  // 24 KB
  const int tid = threadIdx.x;
  const int wave = tid >> 5, lane = tid & 31;
  const int mBase = blockIdx.x * 16;
  const float* Xrow = X + (size_t)mBase * kC;  // 16 contiguous rows
  for (int e = tid; e < 16 * kC; e += 256) xa[e] = (_Float16)Xrow[e];
  __syncthreads();
  for (int nt = wave; nt < kC3 / 16; nt += 8) {
    const int nBase = nt * 16;
    if (nt + 8 < kC3 / 16)
      __builtin_prefetch(WqkvT + (size_t)((nt + 8) * 16 + (lane & 15)) * kC, 0, 3);
    v8f acc = {};
    for (int ks = 0; ks < kC; ks += 32) {
      v16h a = load_a_h(xa, kC, 0, ks, lane);
      v16h b = load_bth(WqkvT, kC, ks, nBase, lane);
      acc = wmma_f16(a, b, acc);
    }
    store_c(O, kC3, mBase, nBase, lane, acc);
  }
}

// ============ 2) orthogonal landmark selection (one block per bh) ============
__global__ __launch_bounds__(256) void k_landmarks(const float* __restrict__ qkv,
                                                   _Float16* __restrict__ lmh,
                                                   IdxArr idx0) {
  __shared__ float msim[kN];
  __shared__ float qn[kN];
  __shared__ float cur[kD];
  __shared__ float redv[256];
  __shared__ int   redi[256];
  __shared__ int   idxl[kL];
  const int bh = blockIdx.x;
  const int b = bh / kH, hh = bh % kH;
  const int tid = threadIdx.x;
  const float* qb = qkv + (size_t)b * kN * kC3 + hh * kD;
  for (int n = tid; n < kN; n += 256) {
    const float* row = qb + (size_t)n * kC3;
    float s = 0.f;
#pragma unroll 8
    for (int j = 0; j < kD; ++j) { float v = row[j]; s += v * v; }
    qn[n] = 1.0f / fmaxf(sqrtf(s), 1e-12f);  // scale cancels in cosines
    msim[n] = 0.f;
  }
  __syncthreads();
  if (tid == 0) { idxl[0] = idx0.v[bh]; msim[idx0.v[bh]] = 10.0f; }
  __syncthreads();
  {
    int i0 = idxl[0];
    if (tid < kD) cur[tid] = qb[(size_t)i0 * kC3 + tid] * qn[i0];
  }
  __syncthreads();
  for (int it = 1; it < kL; ++it) {
    float bestv = 1e30f; int besti = 0x7fffffff;
    for (int n = tid; n < kN; n += 256) {
      const float* row = qb + (size_t)n * kC3;
      float dt = 0.f;
#pragma unroll 8
      for (int j = 0; j < kD; ++j) dt += row[j] * cur[j];
      float cs = fabsf(dt * qn[n]);
      float m = fmaxf(msim[n], cs);
      msim[n] = m;
      if (m < bestv) { bestv = m; besti = n; }
    }
    redv[tid] = bestv; redi[tid] = besti;
    __syncthreads();
    for (int s = 128; s > 0; s >>= 1) {
      if (tid < s) {
        float v2 = redv[tid + s]; int i2 = redi[tid + s];
        if (v2 < redv[tid] || (v2 == redv[tid] && i2 < redi[tid])) {
          redv[tid] = v2; redi[tid] = i2;
        }
      }
      __syncthreads();
    }
    if (tid == 0) { int ix = redi[0]; idxl[it] = ix; msim[ix] = 10.0f; }
    __syncthreads();
    {
      int ix = idxl[it];
      if (tid < kD) cur[tid] = qb[(size_t)ix * kC3 + tid] * qn[ix];
    }
    __syncthreads();
  }
  for (int e = tid; e < kL * kD; e += 256) {  // landmarks = scaled q rows (f16)
    int i = e >> 6, j = e & 63;
    lmh[((size_t)bh * kL + i) * kD + j] =
        (_Float16)(qb[(size_t)idxl[i] * kC3 + j] * kScQK);
  }
}

// ============ 3) kernel_1 scores: q (N x 64) @ lm^T (64 x 128) ============
__global__ __launch_bounds__(256) void k_k1_scores(const float* __restrict__ qkv,
                                                   const _Float16* __restrict__ lmh,
                                                   float* __restrict__ k1w) {
  const int wave = threadIdx.x >> 5, lane = threadIdx.x & 31;
  const int bh = blockIdx.y, b = bh / kH, hh = bh % kH;
  const int mBase = blockIdx.x * 16;
  const int nBase = wave * 16;  // 8 waves cover L=128
  const float* A = qkv + (size_t)b * kN * kC3 + hh * kD;
  const _Float16* Bt = lmh + (size_t)bh * kL * kD;  // lm rows are Bt layout
  v8f acc = {};
  for (int ks = 0; ks < kD; ks += 32) {
    v16h a = load_a_f32v(A, kC3, mBase, ks, lane, kScQK);
    v16h bf = load_bth(Bt, kD, ks, nBase, lane);
    acc = wmma_f16(a, bf, acc);
  }
  store_c(k1w + (size_t)bh * kN * kL, kL, mBase, nBase, lane, acc);
}

// ============ 4) softmax over 128, f32 in -> f16 out ============
__global__ __launch_bounds__(256) void k_softmax128(const float* __restrict__ src,
                                                    _Float16* __restrict__ dst) {
  const int wave = threadIdx.x >> 5, lane = threadIdx.x & 31;
  const size_t r = (size_t)blockIdx.x * 8 + wave;
  const float* row = src + r * kL;
  _Float16* out = dst + r * kL;
  float v[4]; float mx = -1e30f;
#pragma unroll
  for (int i = 0; i < 4; ++i) { v[i] = row[lane + 32 * i]; mx = fmaxf(mx, v[i]); }
  mx = wred_max(mx);
  float s = 0.f;
#pragma unroll
  for (int i = 0; i < 4; ++i) { v[i] = expf(v[i] - mx); s += v[i]; }
  s = wred_sum(s);
  float inv = 1.0f / s;
#pragma unroll
  for (int i = 0; i < 4; ++i) out[lane + 32 * i] = (_Float16)(v[i] * inv);
}

// ============ 5) kernel_2 scores: lm (128x64) @ k^T (64 x N) ============
__global__ __launch_bounds__(256) void k_k2_scores(const float* __restrict__ qkv,
                                                   const _Float16* __restrict__ lmh,
                                                   float* __restrict__ k2w) {
  const int wave = threadIdx.x >> 5, lane = threadIdx.x & 31;
  const int bh = blockIdx.y, b = bh / kH, hh = bh % kH;
  const int mBase = wave * 16;        // 8 waves cover L=128
  const int nBase = blockIdx.x * 16;  // over N
  const _Float16* A = lmh + (size_t)bh * kL * kD;
  const float* Km = qkv + (size_t)b * kN * kC3 + kC + hh * kD;  // k rows
  v8f acc = {};
  for (int ks = 0; ks < kD; ks += 32) {
    v16h a = load_a_h(A, kD, mBase, ks, lane);
    v16h bf = load_bt_f32v(Km, kC3, ks, nBase, lane, kScQK);
    acc = wmma_f16(a, bf, acc);
  }
  store_c(k2w + (size_t)bh * kL * kN, kN, mBase, nBase, lane, acc);
}

// ============ 6) per-frame softmax over P=196, f32 in -> f16 out ============
__global__ __launch_bounds__(256) void k_softmax196(const float* __restrict__ src,
                                                    _Float16* __restrict__ dst) {
  const int wave = threadIdx.x >> 5, lane = threadIdx.x & 31;
  const int r = blockIdx.x * 8 + wave;  // (bh*L + l)*16 + f
  const size_t off = (size_t)(r >> 4) * kN + (r & 15) * kP;
  const float* row = src + off;
  _Float16* out = dst + off;
  float v[7]; float mx = -1e30f;
#pragma unroll
  for (int i = 0; i < 7; ++i) {
    int p = lane + 32 * i;
    v[i] = (p < kP) ? row[p] : -1e30f;
    mx = fmaxf(mx, v[i]);
  }
  mx = wred_max(mx);
  float s = 0.f;
#pragma unroll
  for (int i = 0; i < 7; ++i) {
    int p = lane + 32 * i;
    if (p < kP) { v[i] = expf(v[i] - mx); s += v[i]; }
  }
  s = wred_sum(s);
  float inv = 1.0f / s;
#pragma unroll
  for (int i = 0; i < 7; ++i) {
    int p = lane + 32 * i;
    if (p < kP) out[p] = (_Float16)(v[i] * inv);
  }
}

// ===== 7) xlmT[bh, f*64+j, l] = sum_p kernel2[l,f,p] * v[p,j]  (128x196x64) =====
__global__ __launch_bounds__(256) void k_xlm(const _Float16* __restrict__ k2h,
                                             const float* __restrict__ qkv,
                                             _Float16* __restrict__ xlmT) {
  const int wave = threadIdx.x >> 5, lane = threadIdx.x & 31;
  const int f = blockIdx.x;
  const int bh = blockIdx.y, b = bh / kH, hh = bh % kH;
  const int mBase = wave * 16;  // 8 waves cover L=128
  const _Float16* Ak = k2h + (size_t)bh * kL * kN + f * kP;  // [l][p], lda=kN
  const float* Bm = qkv + (size_t)(b * kN + f * kP) * kC3 + 2 * kC + hh * kD;
  for (int nt = 0; nt < 4; ++nt) {
    const int nBase = nt * 16;
    v8f acc = {};
    for (int ks = 0; ks < 192; ks += 32) {  // K 0..191 unchecked
      v16h a = load_a_hs<false>(Ak, kN, mBase, ks, lane, kP);
      v16h bf = load_b<false>(Bm, kC3, ks, nBase, lane, 1.0f, kP);
      acc = wmma_f16(a, bf, acc);
    }
    {  // K tail 192..195
      v16h a = load_a_hs<true>(Ak, kN, mBase, 192, lane, kP);
      v16h bf = load_b<true>(Bm, kC3, 192, nBase, lane, 1.0f, kP);
      acc = wmma_f16(a, bf, acc);
    }
    const int jc = f * kD + nBase + (lane & 15);  // column index in (F*D)
    const int mAdd = (lane >= 16) ? 8 : 0;
#pragma unroll
    for (int v = 0; v < 8; ++v) {
      int m = mBase + v + mAdd;  // landmark l
      xlmT[((size_t)bh * kFD + jc) * kL + m] = (_Float16)acc[v];
    }
  }
}

// ===== 8) xo = kernel1 @ xlm : A tile (4KB f16) async-staged to LDS =====
__global__ __launch_bounds__(256) void k_xo(const _Float16* __restrict__ k1h,
                                            const _Float16* __restrict__ xlmT,
                                            float* __restrict__ xo) {
  __shared__ _Float16 at[16 * kL];  // 4 KB
  const int tid = threadIdx.x;
  const int wave = tid >> 5, lane = tid & 31;
  const int bh = blockIdx.z, b = bh / kH, hh = bh % kH;
  const int mBase = blockIdx.x * 16;
  const _Float16* Ag = k1h + (size_t)(bh * kN + mBase) * kL;  // contiguous 4KB
#if HAVE_ASYNC
  __builtin_amdgcn_global_load_async_to_lds_b128(
      (AS1 v4i*)(Ag + (size_t)tid * 8),
      (AS3 v4i*)(at + (size_t)tid * 8), 0, 0);
  __builtin_amdgcn_s_wait_asynccnt(0);
#else
  ((v8h*)at)[tid] = ((const v8h*)Ag)[tid];
#endif
  __syncthreads();
  const int nBase = (blockIdx.y * 8 + wave) * 16;  // over F*D = 1024
  const _Float16* Bt = xlmT + (size_t)bh * kFD * kL;
  v8f acc = {};
  for (int ks = 0; ks < kL; ks += 32) {
    v16h a = load_a_h(at, kL, 0, ks, lane);
    v16h bf = load_bth(Bt, kL, ks, nBase, lane);
    acc = wmma_f16(a, bf, acc);
  }
  const int col = nBase + (lane & 15);
  const int f = col >> 6, j = col & 63;
  const int mAdd = (lane >= 16) ? 8 : 0;
#pragma unroll
  for (int v = 0; v < 8; ++v) {
    int m = mBase + v + mAdd;  // token n
    xo[(((size_t)(b * kN + m)) * kF + f) * kC + hh * kD + j] = acc[v];
  }
}

// ============ 9) q2 = xdiag @ W_pq : diag rows gathered into LDS ============
__global__ __launch_bounds__(256) void k_q2(const float* __restrict__ xo,
                                            const _Float16* __restrict__ WpqT,
                                            float* __restrict__ q2) {
  __shared__ _Float16 xa[16 * kC];
  const int tid = threadIdx.x;
  const int wave = tid >> 5, lane = tid & 31;
  const int mBase = blockIdx.x * 16;
  for (int e = tid; e < 16 * kC; e += 256) {
    int r = e / kC, c = e - r * kC;
    int m = mBase + r;          // m = b*N + n
    int n = m % kN;
    int fn = n / kP;            // frame-diagonal gather
    xa[e] = (_Float16)xo[((size_t)m * kF + fn) * kC + c];
  }
  __syncthreads();
  for (int nt = wave; nt < kC / 16; nt += 8) {
    const int nBase = nt * 16;
    v8f acc = {};
    for (int ks = 0; ks < kC; ks += 32) {
      v16h a = load_a_h(xa, kC, 0, ks, lane);
      v16h bf = load_bth(WpqT, kC, ks, nBase, lane);
      acc = wmma_f16(a, bf, acc);
    }
    store_c(q2, kC, mBase, nBase, lane, acc);
  }
}

// ===== 10) fused: kv = xo_row @ W_pkv (LDS) + per-frame attention =====
__global__ __launch_bounds__(256) void k_fused_attn(const float* __restrict__ xo,
                                                    const float* __restrict__ q2,
                                                    const _Float16* __restrict__ WpkvT,
                                                    float* __restrict__ ctx,
                                                    float* __restrict__ attnOut) {
  __shared__ _Float16 xa[kF * kC];   // 24 KB
  __shared__ _Float16 kv[kF][kC2];   // 48 KB: [k2 | v2]
  __shared__ float    q2s[kC];
  __shared__ float    lgt[kH][kF];
  __shared__ float    aw[kH][kF];
  const int tid = threadIdx.x;
  const int wave = tid >> 5, lane = tid & 31;
  const int gid = blockIdx.x;  // b*N + n
  const int b = gid / kN, n = gid % kN;
  const float* xrow = xo + (size_t)gid * kF * kC;
  for (int e = tid; e < kF * kC; e += 256) xa[e] = (_Float16)xrow[e];
  for (int e = tid; e < kC; e += 256) q2s[e] = q2[(size_t)gid * kC + e];
  __syncthreads();
  // kv[16][1536] = xa[16][768] @ W_pkv ; 8 waves x 12 col tiles
  for (int nt = 0; nt < 12; ++nt) {
    const int colBase = (wave * 12 + nt) * 16;
    if (nt + 1 < 12)
      __builtin_prefetch(WpkvT + (size_t)(colBase + 16 + (lane & 15)) * kC, 0, 3);
    v8f acc = {};
    for (int ks = 0; ks < kC; ks += 32) {
      v16h a = load_a_h(xa, kC, 0, ks, lane);
      v16h bf = load_bth(WpkvT, kC, ks, colBase, lane);
      acc = wmma_f16(a, bf, acc);
    }
    const int col = colBase + (lane & 15);
    const int mAdd = (lane >= 16) ? 8 : 0;
#pragma unroll
    for (int v = 0; v < 8; ++v) kv[v + mAdd][col] = (_Float16)acc[v];
  }
  __syncthreads();
  if (tid < kH * kF) {  // logits[h][f] = scale * q2[h,:].k2[f,h,:]
    const int hh = tid >> 4, f = tid & 15;
    float s = 0.f;
#pragma unroll 8
    for (int j = 0; j < kD; ++j) s += q2s[hh * kD + j] * (float)kv[f][hh * kD + j];
    lgt[hh][f] = s * kScAt;
  }
  __syncthreads();
  if (tid < kH) {
    const int hh = tid;
    float mx = -1e30f;
    for (int f = 0; f < kF; ++f) mx = fmaxf(mx, lgt[hh][f]);
    float e[kF]; float sum = 0.f;
    for (int f = 0; f < kF; ++f) { e[f] = expf(lgt[hh][f] - mx); sum += e[f]; }
    float inv = 1.0f / sum;
    for (int f = 0; f < kF; ++f) {
      float a = e[f] * inv;
      aw[hh][f] = a;
      attnOut[(((size_t)(b * kH + hh)) * kN + n) * kF + f] = a;
    }
  }
  __syncthreads();
  for (int c = tid; c < kC; c += 256) {  // ctx = attn @ v2
    const int hh = c >> 6;
    float s = 0.f;
#pragma unroll
    for (int f = 0; f < kF; ++f) s += aw[hh][f] * (float)kv[f][kC + c];
    ctx[(size_t)gid * kC + c] = s;
  }
}

// ============ 11) out = ctx @ W_proj + b_proj ============
__global__ __launch_bounds__(256) void k_out(const float* __restrict__ ctx,
                                             const _Float16* __restrict__ WprojT,
                                             const float* __restrict__ bias,
                                             float* __restrict__ out) {
  __shared__ _Float16 xa[16 * kC];
  const int tid = threadIdx.x;
  const int wave = tid >> 5, lane = tid & 31;
  const int mBase = blockIdx.x * 16;
  const float* Crow = ctx + (size_t)mBase * kC;
  for (int e = tid; e < 16 * kC; e += 256) xa[e] = (_Float16)Crow[e];
  __syncthreads();
  for (int nt = wave; nt < kC / 16; nt += 8) {
    const int nBase = nt * 16;
    v8f acc = {};
    for (int ks = 0; ks < kC; ks += 32) {
      v16h a = load_a_h(xa, kC, 0, ks, lane);
      v16h bf = load_bth(WprojT, kC, ks, nBase, lane);
      acc = wmma_f16(a, bf, acc);
    }
    const int col = nBase + (lane & 15);
    const float bb = bias[col];
    const int mAdd = (lane >= 16) ? 8 : 0;
#pragma unroll
    for (int v = 0; v < 8; ++v)
      out[(size_t)(mBase + v + mAdd) * kC + col] = acc[v] + bb;
  }
}

// =========================== host launcher ===========================
extern "C" void kernel_launch(void* const* d_in, const int* in_sizes, int n_in,
                              void* d_out, int out_size, void* d_ws,
                              size_t ws_size, hipStream_t stream) {
  const float* x      = (const float*)d_in[0];
  const float* W_qkv  = (const float*)d_in[1];
  const float* W_pq   = (const float*)d_in[2];
  const float* W_pkv  = (const float*)d_in[3];
  const float* W_proj = (const float*)d_in[4];
  const float* b_proj = (const float*)d_in[5];
  float* out  = (float*)d_out;
  float* attn = out + (size_t)kB * kN * kC;

  float* ws = (float*)d_ws;
  size_t o = 0;
  float* qkv = ws + o;                     o += (size_t)kM * kC3;
  float* k1w = ws + o;                     o += (size_t)kBH * kN * kL;
  float* k2w = ws + o;                     o += (size_t)kBH * kL * kN;
  float* xo  = ws + o;                     o += (size_t)kB * kN * kF * kC;
  float* q2  = ws + o;                     o += (size_t)kB * kN * kC;
  float* ctx = ws + o;                     o += (size_t)kB * kN * kC;
  _Float16* lmh    = (_Float16*)(ws + o);  o += (size_t)kBH * kL * kD / 2;
  _Float16* k1h    = (_Float16*)(ws + o);  o += (size_t)kBH * kN * kL / 2;
  _Float16* k2h    = (_Float16*)(ws + o);  o += (size_t)kBH * kL * kN / 2;
  _Float16* xlmT   = (_Float16*)(ws + o);  o += (size_t)kBH * kFD * kL / 2;
  _Float16* WqkvT  = (_Float16*)(ws + o);  o += (size_t)kC * kC3 / 2;
  _Float16* WpqT   = (_Float16*)(ws + o);  o += (size_t)kC * kC / 2;
  _Float16* WpkvT  = (_Float16*)(ws + o);  o += (size_t)kC * kC2 / 2;
  _Float16* WprojT = (_Float16*)(ws + o);  o += (size_t)kC * kC / 2;

  // Host-side Threefry-2x32 replicating jax.random.randint(key(42),(24,),0,N).
  IdxArr idx0;
  {
    auto rotl = [](uint32_t v, int r) { return (v << r) | (v >> (32 - r)); };
    const uint32_t key0 = 0u, key1 = 42u;
    const uint32_t ksv[3] = {key0, key1, key0 ^ key1 ^ 0x1BD11BDAu};
    const int rots[2][4] = {{13, 15, 26, 6}, {17, 29, 16, 24}};
    uint32_t bits[24];
    for (int i = 0; i < 12; ++i) {
      uint32_t x0 = (uint32_t)i + key0, x1 = (uint32_t)(i + 12) + key1;
      for (int g = 0; g < 5; ++g) {
        const int* rr = rots[g & 1];
        for (int j = 0; j < 4; ++j) { x0 += x1; x1 = rotl(x1, rr[j]); x1 ^= x0; }
        x0 += ksv[(g + 1) % 3];
        x1 += ksv[(g + 2) % 3] + (uint32_t)(g + 1);
      }
      bits[i] = x0; bits[i + 12] = x1;
    }
    const uint32_t span = (uint32_t)kN;
    uint32_t mlt = 65536u % span;
    mlt = (uint32_t)(((uint64_t)mlt * mlt) % span);
    for (int i = 0; i < 24; ++i) {
      uint32_t hi = bits[i] >> 16, lo = bits[i] & 0xFFFFu;
      idx0.v[i] = (int)((((uint64_t)(hi % span)) * mlt + (lo % span)) % span);
    }
  }

  const dim3 blk(256);
  k_wt_cvt<<<dim3((kC * kC3 + 255) / 256), blk, 0, stream>>>(W_qkv, WqkvT, kC, kC3);
  k_wt_cvt<<<dim3((kC * kC  + 255) / 256), blk, 0, stream>>>(W_pq, WpqT, kC, kC);
  k_wt_cvt<<<dim3((kC * kC2 + 255) / 256), blk, 0, stream>>>(W_pkv, WpkvT, kC, kC2);
  k_wt_cvt<<<dim3((kC * kC  + 255) / 256), blk, 0, stream>>>(W_proj, WprojT, kC, kC);

  k_gemm_qkv  <<<dim3(kM / 16),            blk, 0, stream>>>(x, WqkvT, qkv);
  k_landmarks <<<dim3(kBH),                blk, 0, stream>>>(qkv, lmh, idx0);
  k_k1_scores <<<dim3(kN / 16, kBH),       blk, 0, stream>>>(qkv, lmh, k1w);
  k_softmax128<<<dim3(kBH * kN / 8),       blk, 0, stream>>>(k1w, k1h);
  k_k2_scores <<<dim3(kN / 16, kBH),       blk, 0, stream>>>(qkv, lmh, k2w);
  k_softmax196<<<dim3(kBH * kL * kF / 8),  blk, 0, stream>>>(k2w, k2h);
  k_xlm       <<<dim3(kF, kBH),            blk, 0, stream>>>(k2h, qkv, xlmT);
  k_xo        <<<dim3(kN / 16, 8, kBH),    blk, 0, stream>>>(k1h, xlmT, xo);
  k_q2        <<<dim3(kM / 16),            blk, 0, stream>>>(xo, WpqT, q2);
  k_fused_attn<<<dim3(kB * kN),            blk, 0, stream>>>(xo, q2, WpkvT, ctx, attn);
  k_out       <<<dim3(kM / 16),            blk, 0, stream>>>(ctx, WprojT, b_proj, out);

  (void)in_sizes; (void)n_in; (void)out_size; (void)ws_size; (void)o;
}
